// SelfAttention_12695923327315
// MI455X (gfx1250) — compile-verified, matrix operand order checked by reading, not run on hardware
//
#include <hip/hip_runtime.h>

typedef __bf16 bf16_t;
typedef __attribute__((ext_vector_type(16))) __bf16 v16bf;
typedef __attribute__((ext_vector_type(8)))  __bf16 v8bf;
typedef __attribute__((ext_vector_type(8)))  float   v8f;
typedef unsigned int v4u_t __attribute__((ext_vector_type(4)));
typedef int          v8i_t __attribute__((ext_vector_type(8)));
typedef int          v4i_t __attribute__((ext_vector_type(4)));

#define FEAT   1024
#define SEQ    2048
#define NBATCH 8
#define ROWS   (NBATCH * SEQ)

#define BQ   128   // queries per workgroup (16 per wave)
#define BK   64    // key block
#define DSL  128   // feature slice per workgroup for the O accumulator
#define VSTR 72    // padded LDS row stride for Vt/P (bf16): 144B, 16B-aligned
// Q tile LDS layout matches TDM padding: 8 bf16 inserted every 512 bf16.
// Effective row stride = 1024 + 2*8 = 1040 bf16 (2080B -> 8-bank rotation).
#define QSTR 1040

#define SMEM_ELEMS (BQ * QSTR + DSL * VSTR + 8 * 16 * VSTR)

#if __has_builtin(__builtin_amdgcn_tensor_load_to_lds) && \
    __has_builtin(__builtin_amdgcn_s_wait_tensorcnt)
#define HAVE_TDM 1
#else
#define HAVE_TDM 0
#endif

__device__ __forceinline__ v8f vzero8() {
  v8f z;
#pragma unroll
  for (int i = 0; i < 8; ++i) z[i] = 0.0f;
  return z;
}

// Build a 16x32 bf16 WMMA A/B fragment from a row-major source (global or LDS).
// Lane L holds row (row0 + L%16). Element j holds K index
//   k0 + (L/16)*8 + (j<8 ? j : (j-8)+16)
// per the CDNA5 16-bit A-matrix VGPR layout (B is the symmetric layout, so
// "rows" of the source are B's columns). Two contiguous 16B loads.
__device__ __forceinline__ v16bf load_frag(const bf16_t* __restrict__ base,
                                           int row0, int ld, int k0) {
  const int lane = threadIdx.x & 31;
  const int g    = lane >> 4;
  const int r    = lane & 15;
  const bf16_t* p = base + (size_t)(row0 + r) * ld + (k0 + g * 8);
  v8bf lo = *(const v8bf*)(p);
  v8bf hi = *(const v8bf*)(p + 16);
  v16bf a;
#pragma unroll
  for (int i = 0; i < 8; ++i) { a[i] = lo[i]; a[8 + i] = hi[i]; }
  return a;
}

__device__ __forceinline__ v8f wmma_bf16(v16bf a, v16bf b, v8f c) {
  return __builtin_amdgcn_wmma_f32_16x16x32_bf16(false, a, false, b,
                                                 (short)0, c, false, false);
}

// reductions across the 16-lane half groups (C-frag columns live there)
__device__ __forceinline__ float redmax16(float v) {
  v = fmaxf(v, __shfl_xor(v, 1, 32));
  v = fmaxf(v, __shfl_xor(v, 2, 32));
  v = fmaxf(v, __shfl_xor(v, 4, 32));
  v = fmaxf(v, __shfl_xor(v, 8, 32));
  return v;
}
__device__ __forceinline__ float redsum16(float v) {
  v += __shfl_xor(v, 1, 32);
  v += __shfl_xor(v, 2, 32);
  v += __shfl_xor(v, 4, 32);
  v += __shfl_xor(v, 8, 32);
  return v;
}

// ---------------- kernel 1: fp32 -> bf16 ----------------
__global__ void cvt_bf16_kernel(const float* __restrict__ src,
                                bf16_t* __restrict__ dst, int n) {
  for (int i = blockIdx.x * blockDim.x + threadIdx.x; i < n;
       i += gridDim.x * blockDim.x)
    dst[i] = (bf16_t)src[i];
}

// ---------------- kernel 2: Out = X * W^T + bias (bf16 in, bf16 out) -------
// 128x128 tile per workgroup, 8 waves as 4(M) x 2(N), each wave 32x64.
// Fragments come straight from global: all operands are L2-resident (192MB L2
// holds the full working set) and intra-WG fragment redundancy hits the WGP$.
__global__ __launch_bounds__(256) void linear_gemm_kernel(
    const bf16_t* __restrict__ X, const bf16_t* __restrict__ W,
    const float* __restrict__ bias, bf16_t* __restrict__ Out) {
  const int wave = threadIdx.x >> 5;
  const int wm = blockIdx.x * 128 + (wave & 3) * 32;
  const int wn = blockIdx.y * 128 + (wave >> 2) * 64;
  const int lane = threadIdx.x & 31;
  const int hi = lane >> 4, lc = lane & 15;

  v8f c[2][4];
#pragma unroll
  for (int i = 0; i < 2; ++i)
#pragma unroll
    for (int j = 0; j < 4; ++j) c[i][j] = vzero8();

  for (int k0 = 0; k0 < FEAT; k0 += 32) {
    v16bf a0 = load_frag(X, wm,      FEAT, k0);
    v16bf a1 = load_frag(X, wm + 16, FEAT, k0);
#pragma unroll
    for (int nf = 0; nf < 4; ++nf) {
      v16bf b = load_frag(W, wn + nf * 16, FEAT, k0);  // B[k=d][n=e] = W[e][d]
      c[0][nf] = wmma_bf16(a0, b, c[0][nf]);
      c[1][nf] = wmma_bf16(a1, b, c[1][nf]);
    }
  }
#pragma unroll
  for (int nf = 0; nf < 4; ++nf) {
    float bv = bias[wn + nf * 16 + lc];
#pragma unroll
    for (int mi = 0; mi < 2; ++mi)
#pragma unroll
      for (int r = 0; r < 8; ++r) {
        int row = wm + mi * 16 + r + hi * 8;
        Out[(size_t)row * FEAT + wn + nf * 16 + lc] =
            (bf16_t)(c[mi][nf][r] + bv);
      }
  }
}

// ---------------- kernel 3: flash attention ----------------
// grid: (SEQ/BQ, FEAT/DSL, NBATCH); block 256 (8 waves, 16 queries each).
// The full 128x1024 bf16 Q tile (~260KB padded) lives in LDS for the whole
// kernel — only possible with CDNA5's 320KB LDS/WGP. Staged by the Tensor
// Data Mover when available (TENSORcnt + s_wait_tensorcnt), else manually.
__global__ __launch_bounds__(256) void attn_kernel(
    const bf16_t* __restrict__ Q, const bf16_t* __restrict__ K,
    const bf16_t* __restrict__ V, float* __restrict__ Out) {
  extern __shared__ __align__(16) bf16_t smem[];
  bf16_t* Qs = smem;                       // [BQ][QSTR]   staged Q tile
  bf16_t* Vt = Qs + BQ * QSTR;             // [DSL][VSTR]  V^T tile
  bf16_t* Pl = Vt + DSL * VSTR;            // 8 x [16][VSTR] per-wave P tile

  const int qTile  = blockIdx.x * BQ;
  const int dsBase = blockIdx.y * DSL;
  const int bidx   = blockIdx.z;
  const int wave   = threadIdx.x >> 5;
  const int lane   = threadIdx.x & 31;
  const int hi = lane >> 4, lc = lane & 15;
  const size_t rowBase = (size_t)bidx * SEQ;
  const bf16_t* Kb = K + rowBase * FEAT;
  const float scale = 0.03125f;            // 1/sqrt(1024)

  // ---- stage the whole Q tile into LDS (128 rows x 1024 bf16, 8-elem pad
  //      inserted every 512 elements -> effective row stride QSTR=1040)
  const bf16_t* Qg = Q + (rowBase + qTile) * FEAT;
#if HAVE_TDM
  if (wave == 0) {
    unsigned long long ga = (unsigned long long)(size_t)Qg;
    v4u_t g0;
    g0[0] = 1u;                                   // count=1 (valid, user mode)
    g0[1] = 0u;                                   // lds_addr: dynamic LDS base
    g0[2] = (unsigned)(ga & 0xFFFFFFFFu);         // global_addr[31:0]
    g0[3] = (unsigned)((ga >> 32) & 0x01FFFFFFu)  // global_addr[56:32]
            | (2u << 30);                         // type = 2 ("image")
    v8i_t g1;
    g1[0] = (1 << 16)       // data_size = 2 bytes
          | (1 << 20)       // pad_enable
          | (7 << 22)       // pad_interval: 256 DWORDs = 512 bf16
          | (3 << 25);      // pad_amount: 4 DWORDs = 8 bf16
    g1[1] = (int)(1024u << 16);   // tensor_dim0 = 1024 (bits 79:48)
    g1[2] = (int)(128u  << 16);   // tensor_dim1 = 128  (bits 111:80)
    g1[3] = (int)(1024u << 16);   // tile_dim0   = 1024 (bits 127:112)
    g1[4] = 128;                  // tile_dim1   = 128  (bits 143:128)
    g1[5] = 1024;                 // tensor_dim0_stride = 1024 (bits 207:160)
    g1[6] = 0;
    g1[7] = 0;
    v4i_t gz;
    gz[0] = 0; gz[1] = 0; gz[2] = 0; gz[3] = 0;
#if defined(__clang_major__) && (__clang_major__ >= 23)
    v8i_t g4;
#pragma unroll
    for (int i = 0; i < 8; ++i) g4[i] = 0;
    __builtin_amdgcn_tensor_load_to_lds(g0, g1, gz, gz, g4, 0);
#else
    __builtin_amdgcn_tensor_load_to_lds(g0, g1, gz, gz, 0);
#endif
    __builtin_amdgcn_s_wait_tensorcnt((short)0);
  }
#else
#pragma unroll 4
  for (int it = 0; it < 64; ++it) {
    int c   = threadIdx.x + it * 256;      // 16384 chunks of 8 bf16
    int row = c >> 7;
    int col = (c & 127) * 8;
    *(v8bf*)(Qs + row * QSTR + col + (col >= 512 ? 8 : 0)) =
        *(const v8bf*)(Qg + (size_t)row * FEAT + col);
  }
#endif
  __syncthreads();

  v8f o[8];
#pragma unroll
  for (int i = 0; i < 8; ++i) o[i] = vzero8();
  float mrow[8], lrow[8];
#pragma unroll
  for (int r = 0; r < 8; ++r) { mrow[r] = -1.0e30f; lrow[r] = 0.0f; }

  bf16_t* pw = Pl + wave * 16 * VSTR;

  for (int kb = 0; kb < SEQ; kb += BK) {
    __syncthreads();  // previous iteration's LDS consumers done
    // ---- prefetch next key block (lowers to global_prefetch_b8)
    if (kb + BK < SEQ) {
      __builtin_prefetch(Kb + (size_t)(kb + BK + lane) * FEAT, 0, 1);
      __builtin_prefetch(Kb + (size_t)(kb + BK + 32 + lane) * FEAT, 0, 1);
      __builtin_prefetch(V + (rowBase + kb + BK + lane) * FEAT + dsBase, 0, 1);
      __builtin_prefetch(V + (rowBase + kb + BK + 32 + lane) * FEAT + dsBase, 0, 1);
    }
    // ---- cooperatively stage V^T slice: Vt[d][key] = V[kb+key][dsBase+d]
#pragma unroll
    for (int it = 0; it < 4; ++it) {
      int chunk = threadIdx.x + it * 256;    // 0..1023, 8 bf16 each
      int key = chunk >> 4;
      int d8  = (chunk & 15) * 8;
      v8bf vv = *(const v8bf*)(V + (rowBase + kb + key) * FEAT + dsBase + d8);
#pragma unroll
      for (int j = 0; j < 8; ++j) Vt[(d8 + j) * VSTR + key] = vv[j];
    }
    // ---- S = Q * K^T over full FEAT (Q frags from LDS, K frags from L2/L0)
    v8f s[4];
#pragma unroll
    for (int i = 0; i < 4; ++i) s[i] = vzero8();
    for (int k0 = 0; k0 < FEAT; k0 += 32) {
      int kq = k0 + (k0 >= 512 ? 8 : 0);     // skip mid-row pad
      v16bf a = load_frag(Qs, wave * 16, QSTR, kq);
#pragma unroll
      for (int nf = 0; nf < 4; ++nf) {
        v16bf b = load_frag(Kb, kb + nf * 16, FEAT, k0);  // B[k=d][n=key]
        s[nf] = wmma_bf16(a, b, s[nf]);
      }
    }
    // ---- online softmax update (row = r + 8*hi, cols across 16-lane group)
    float mnew[8], alpha[8];
#pragma unroll
    for (int nf = 0; nf < 4; ++nf)
#pragma unroll
      for (int r = 0; r < 8; ++r) s[nf][r] *= scale;
#pragma unroll
    for (int r = 0; r < 8; ++r) {
      float v = fmaxf(fmaxf(s[0][r], s[1][r]), fmaxf(s[2][r], s[3][r]));
      v = redmax16(v);
      mnew[r]  = fmaxf(mrow[r], v);
      alpha[r] = __expf(mrow[r] - mnew[r]);
      mrow[r]  = mnew[r];
    }
#pragma unroll
    for (int nf = 0; nf < 4; ++nf)
#pragma unroll
      for (int r = 0; r < 8; ++r) s[nf][r] = __expf(s[nf][r] - mnew[r]);
#pragma unroll
    for (int r = 0; r < 8; ++r) {
      float t = s[0][r] + s[1][r] + s[2][r] + s[3][r];
      t = redsum16(t);
      lrow[r] = lrow[r] * alpha[r] + t;
    }
#pragma unroll
    for (int i = 0; i < 8; ++i)
#pragma unroll
      for (int r = 0; r < 8; ++r) o[i][r] *= alpha[r];
    // ---- P (C-layout) -> LDS (row-major) for A-frag reload
#pragma unroll
    for (int nf = 0; nf < 4; ++nf)
#pragma unroll
      for (int r = 0; r < 8; ++r)
        pw[(r + hi * 8) * VSTR + nf * 16 + lc] = (bf16_t)s[nf][r];
    __syncthreads();  // Vt staged + P visible
    // ---- O += P * V  (K=64 keys in two 32-key steps, N = 128 d-slice)
#pragma unroll
    for (int ks = 0; ks < 2; ++ks) {
      v16bf a = load_frag(pw, 0, VSTR, ks * 32);
#pragma unroll
      for (int nf = 0; nf < 8; ++nf) {
        v16bf b = load_frag(Vt, nf * 16, VSTR, ks * 32);  // B[k=key][n=d]
        o[nf] = wmma_bf16(a, b, o[nf]);
      }
    }
  }
  // ---- epilogue: normalize and store fp32
#pragma unroll
  for (int nf = 0; nf < 8; ++nf)
#pragma unroll
    for (int r = 0; r < 8; ++r) {
      size_t row = rowBase + qTile + wave * 16 + r + hi * 8;
      Out[row * FEAT + dsBase + nf * 16 + lc] = o[nf][r] / lrow[r];
    }
}

// ---------------- host ----------------
extern "C" void kernel_launch(void* const* d_in, const int* in_sizes, int n_in,
                              void* d_out, int out_size, void* d_ws, size_t ws_size,
                              hipStream_t stream) {
  (void)in_sizes; (void)n_in; (void)out_size; (void)ws_size;
  const float* x  = (const float*)d_in[0];
  const float* Wq = (const float*)d_in[1];
  const float* bq = (const float*)d_in[2];
  const float* Wk = (const float*)d_in[3];
  const float* bk = (const float*)d_in[4];
  const float* Wv = (const float*)d_in[5];
  const float* bv = (const float*)d_in[6];
  float* out = (float*)d_out;

  bf16_t* p   = (bf16_t*)d_ws;
  bf16_t* xb  = p; p += (size_t)ROWS * FEAT;
  bf16_t* Wqb = p; p += (size_t)FEAT * FEAT;
  bf16_t* Wkb = p; p += (size_t)FEAT * FEAT;
  bf16_t* Wvb = p; p += (size_t)FEAT * FEAT;
  bf16_t* Qb  = p; p += (size_t)ROWS * FEAT;
  bf16_t* Kb  = p; p += (size_t)ROWS * FEAT;
  bf16_t* Vb  = p; p += (size_t)ROWS * FEAT;

  cvt_bf16_kernel<<<2048, 256, 0, stream>>>(x,  xb,  ROWS * FEAT);
  cvt_bf16_kernel<<<512,  256, 0, stream>>>(Wq, Wqb, FEAT * FEAT);
  cvt_bf16_kernel<<<512,  256, 0, stream>>>(Wk, Wkb, FEAT * FEAT);
  cvt_bf16_kernel<<<512,  256, 0, stream>>>(Wv, Wvb, FEAT * FEAT);

  dim3 gg(ROWS / 128, FEAT / 128, 1);
  linear_gemm_kernel<<<gg, 256, 0, stream>>>(xb, Wqb, bq, Qb);
  linear_gemm_kernel<<<gg, 256, 0, stream>>>(xb, Wkb, bk, Kb);
  linear_gemm_kernel<<<gg, 256, 0, stream>>>(xb, Wvb, bv, Vb);

  const size_t smem_bytes = (size_t)SMEM_ELEMS * sizeof(bf16_t);  // ~296KB
  (void)hipFuncSetAttribute((const void*)attn_kernel,
                            hipFuncAttributeMaxDynamicSharedMemorySize,
                            (int)smem_bytes);
  dim3 ga(SEQ / BQ, FEAT / DSL, NBATCH);
  attn_kernel<<<ga, 256, smem_bytes, stream>>>(Qb, Kb, Vb, out);
}
